// FrozenBNBLinear_14293651161610
// MI455X (gfx1250) — matrix operand order repaired and checked by reading, not compiled, not run on hardware
//
#include <hip/hip_runtime.h>

// ---------------------------------------------------------------------------
// FrozenBNBLinear for MI455X (gfx1250, wave32, WMMA + async global->LDS DMA)
//   out[b,s,o] = sum_i x[b,s,i] * dequant(W)[o,i] + bias[o]
//   M=8192, K=4096, N=16384.  Compute-bound => bf16 WMMA, f32 accumulate.
//   Dequant W once -> bf16 in ws; x -> bf16 in ws (192 MB working set ~ L2).
//   GEMM: 128x128x64 tiles; LDS double-buffer filled by
//   global_load_async_to_lds_b128 in GVS form (SGPR base + 32b lane offset),
//   K-loop unrolled x2 so buffer selection is compile-time (no movrels).
// ---------------------------------------------------------------------------

typedef __bf16 bf16;
typedef __attribute__((ext_vector_type(16))) __bf16 bf16x16;
typedef __attribute__((ext_vector_type(8)))  __bf16 bf16x8;
typedef __attribute__((ext_vector_type(8)))  float  f32x8;

#define IN_F   4096
#define OUT_F  16384
#define MTOT   8192
#define BLKSZ  4096

#define BM 128
#define BN 128
#define BK 64
#define LDSW (BK + 8)            // padded bf16 row stride: 72 elems = 144 B
#define BUFELT (BM * LDSW)       // elements per LDS buffer (A or B)

// ---------------- Phase 1a: x fp32 -> bf16 ---------------------------------
__global__ __launch_bounds__(256)
void cvt_x_kernel(const float* __restrict__ x, bf16* __restrict__ xb) {
  long i = ((long)blockIdx.x * 256 + threadIdx.x) * 8;
  float4 a = *(const float4*)(x + i);
  float4 b = *(const float4*)(x + i + 4);
  bf16x8 o;
  o[0] = (bf16)a.x; o[1] = (bf16)a.y; o[2] = (bf16)a.z; o[3] = (bf16)a.w;
  o[4] = (bf16)b.x; o[5] = (bf16)b.y; o[6] = (bf16)b.z; o[7] = (bf16)b.w;
  *(bf16x8*)(xb + i) = o;
}

// ---------------- Phase 1b: blockwise dequant W -> bf16 --------------------
__global__ __launch_bounds__(256)
void deq_w_kernel(const int* __restrict__ wq, const float* __restrict__ absmax,
                  const float* __restrict__ code, bf16* __restrict__ wb) {
  __shared__ float code_s[256];
  code_s[threadIdx.x] = code[threadIdx.x];
  __syncthreads();

  long i = ((long)blockIdx.x * 256 + threadIdx.x) * 8;
  int4 q0 = *(const int4*)(wq + i);
  int4 q1 = *(const int4*)(wq + i + 4);
  float s = absmax[(long)(i >> 12)];   // BLOCKSIZE=4096, 8-aligned => one block
  bf16x8 o;
  o[0] = (bf16)(code_s[q0.x] * s);
  o[1] = (bf16)(code_s[q0.y] * s);
  o[2] = (bf16)(code_s[q0.z] * s);
  o[3] = (bf16)(code_s[q0.w] * s);
  o[4] = (bf16)(code_s[q1.x] * s);
  o[5] = (bf16)(code_s[q1.y] * s);
  o[6] = (bf16)(code_s[q1.z] * s);
  o[7] = (bf16)(code_s[q1.w] * s);
  *(bf16x8*)(wb + i) = o;
}

// ---- gfx1250 async global->LDS copy, GVS mode: s[base] + v_off + imm ------
template <int IMM>
__device__ __forceinline__ void async_b128_s(unsigned lds_addr, unsigned voff,
                                             const void* sbase) {
  asm volatile("global_load_async_to_lds_b128 %0, %1, %2 offset:%3"
               :: "v"(lds_addr), "v"(voff),
                  "s"((unsigned long long)(uintptr_t)sbase), "i"(IMM)
               : "memory");
}
__device__ __forceinline__ void wait_async0() {
  asm volatile("s_wait_asynccnt 0" ::: "memory");
}

// ---------------- Phase 2: bf16 WMMA GEMM ----------------------------------
__global__ __launch_bounds__(256)
void gemm_bf16_wmma(const bf16* __restrict__ A,   // [MTOT][IN_F]
                    const bf16* __restrict__ B,   // [OUT_F][IN_F]
                    const float* __restrict__ bias,
                    float* __restrict__ C) {      // [MTOT][OUT_F]
  __shared__ bf16 As[2][BUFELT];
  __shared__ bf16 Bs[2][BUFELT];

  // L2-friendly swizzle: groups of 8 n-tiles, sweep all m-tiles inside group.
  const int NT_M = MTOT / BM;   // 64
  const int GRP  = 8;
  int b  = blockIdx.x;
  int g  = b / (GRP * NT_M);
  int r  = b % (GRP * NT_M);
  int tn = g * GRP + (r % GRP);
  int tm = r / GRP;
  const int m0 = tm * BM;
  const int n0 = tn * BN;

  const int t    = threadIdx.x;
  const int lane = t & 31;
  const int wid  = t >> 5;      // 0..7
  const int wm   = wid >> 2;    // 0..1  (64 rows each)
  const int wn   = wid & 3;     // 0..3  (32 cols each)
  const int lrow = lane & 15;
  const int half = lane >> 4;

  // Copy map: tile = 128 rows x 64 bf16 = 8 x 16B chunks/row = 1024 chunks;
  // thread t handles chunks t, t+256, t+512, t+768 of each tile.
  unsigned ldsA0[4], ldsA1[4], ldsB0[4], ldsB1[4];  // loop-invariant LDS addrs
  unsigned voffA[4], voffB[4];                      // byte offsets, k = kp*BK
#pragma unroll
  for (int j = 0; j < 4; ++j) {
    int c   = t + 256 * j;
    int row = c >> 3;
    int col = (c & 7) * 8;
    ldsA0[j] = (unsigned)(uintptr_t)&As[0][row * LDSW + col];
    ldsA1[j] = (unsigned)(uintptr_t)&As[1][row * LDSW + col];
    ldsB0[j] = (unsigned)(uintptr_t)&Bs[0][row * LDSW + col];
    ldsB1[j] = (unsigned)(uintptr_t)&Bs[1][row * LDSW + col];
    voffA[j] = (unsigned)(((m0 + row) * IN_F + col) * 2);
    voffB[j] = (unsigned)(((n0 + row) * IN_F + col) * 2);
  }

  f32x8 acc[4][2];
#pragma unroll
  for (int i = 0; i < 4; ++i)
#pragma unroll
    for (int j = 0; j < 2; ++j)
#pragma unroll
      for (int e = 0; e < 8; ++e) acc[i][j][e] = 0.0f;

  // One 32-K-step slice of the MACs from LDS buffer `buf` at bf16 offset koff.
  auto compute32 = [&](const bf16* Ab, const bf16* Bb, int koff) {
    bf16x16 af[4], bfr[2];
    // A fragment (16-bit A 16x32 layout): lane<16 holds K [0..8)&[16..24),
    // lane>=16 holds K [8..16)&[24..32) of row (lane&15).
#pragma unroll
    for (int i = 0; i < 4; ++i) {
      const bf16* p = Ab + (wm * 64 + i * 16 + lrow) * LDSW + koff;
      bf16x8 lo = *(const bf16x8*)(p + half * 8);
      bf16x8 hi = *(const bf16x8*)(p + 16 + half * 8);
      af[i] = __builtin_shufflevector(lo, hi, 0,1,2,3,4,5,6,7,8,9,10,11,12,13,14,15);
    }
    // B fragment: lane n holds column n (= W row n), same K order as A:
    // lanes 0-15 -> K [0..8)&[16..24), lanes 16-31 -> K [8..16)&[24..32).
#pragma unroll
    for (int i = 0; i < 2; ++i) {
      const bf16* p = Bb + (wn * 32 + i * 16 + lrow) * LDSW + koff;
      bf16x8 lo = *(const bf16x8*)(p + half * 8);
      bf16x8 hi = *(const bf16x8*)(p + 16 + half * 8);
      bfr[i] = __builtin_shufflevector(lo, hi, 0,1,2,3,4,5,6,7,8,9,10,11,12,13,14,15);
    }
#pragma unroll
    for (int i = 0; i < 4; ++i)
#pragma unroll
      for (int j = 0; j < 2; ++j)
        acc[i][j] = __builtin_amdgcn_wmma_f32_16x16x32_bf16(
            false, af[i], false, bfr[j], (short)0, acc[i][j], false, false);
  };

  // Prologue: fill buffer 0 with k = 0.
#pragma unroll
  for (int j = 0; j < 4; ++j) {
    async_b128_s<0>(ldsA0[j], voffA[j], A);
    async_b128_s<0>(ldsB0[j], voffB[j], B);
  }

  const int NKT = IN_F / BK;    // 64 (even)
  for (int kp = 0; kp < NKT; kp += 2) {
    // ---------- half 1: compute buf0, prefetch k=(kp+1)*BK into buf1 -------
    wait_async0();
    __syncthreads();
#pragma unroll
    for (int j = 0; j < 4; ++j) {       // always valid: kp+1 <= 63
      async_b128_s<BK * 2>(ldsA1[j], voffA[j], A);
      async_b128_s<BK * 2>(ldsB1[j], voffB[j], B);
    }
    compute32(As[0], Bs[0], 0);
    compute32(As[0], Bs[0], 32);

    // ---------- half 2: compute buf1, prefetch k=(kp+2)*BK into buf0 -------
    wait_async0();
    __syncthreads();
    if (kp + 2 < NKT) {
#pragma unroll
      for (int j = 0; j < 4; ++j) {
        async_b128_s<2 * BK * 2>(ldsA0[j], voffA[j], A);
        async_b128_s<2 * BK * 2>(ldsB0[j], voffB[j], B);
      }
    }
    compute32(As[1], Bs[1], 0);
    compute32(As[1], Bs[1], 32);

#pragma unroll
    for (int j = 0; j < 4; ++j) { voffA[j] += 2 * BK * 2; voffB[j] += 2 * BK * 2; }
  }

  // Epilogue: C/D 16x16 f32 layout: VGPR e -> (lane<16: M=e,N=lane),
  //                                 (lane>=16: M=e+8,N=lane-16). Fuse bias.
#pragma unroll
  for (int j = 0; j < 2; ++j) {
    int n = n0 + wn * 32 + j * 16 + lrow;
    float bv = bias[n];
#pragma unroll
    for (int i = 0; i < 4; ++i) {
      int mrow = m0 + wm * 64 + i * 16 + half * 8;
#pragma unroll
      for (int e = 0; e < 8; ++e)
        C[(long)(mrow + e) * OUT_F + n] = acc[i][j][e] + bv;
    }
  }
}

// ---------------------------------------------------------------------------
extern "C" void kernel_launch(void* const* d_in, const int* in_sizes, int n_in,
                              void* d_out, int out_size, void* d_ws, size_t ws_size,
                              hipStream_t stream) {
  const float* x      = (const float*)d_in[0];
  const int*   wq     = (const int*)  d_in[1];
  const float* absmax = (const float*)d_in[2];
  const float* code   = (const float*)d_in[3];
  const float* bias   = (const float*)d_in[4];
  float* out = (float*)d_out;

  bf16* xb = (bf16*)d_ws;                                              // 64 MB
  bf16* wb = (bf16*)((char*)d_ws + (size_t)MTOT * IN_F * sizeof(bf16)); // +128 MB

  unsigned gx = (unsigned)(((long)MTOT * IN_F) / 8 / 256);   // 16384 blocks
  unsigned gw = (unsigned)(((long)OUT_F * IN_F) / 8 / 256);  // 32768 blocks
  cvt_x_kernel<<<gx, 256, 0, stream>>>(x, xb);
  deq_w_kernel<<<gw, 256, 0, stream>>>(wq, absmax, code, wb);

  unsigned gg = (unsigned)((MTOT / BM) * (OUT_F / BN));      // 8192 blocks
  gemm_bf16_wmma<<<gg, 256, 0, stream>>>(xb, wb, bias, out);
}
